// CfCHead_3006477107698
// MI455X (gfx1250) — matrix operand order repaired and checked by reading, not compiled, not run on hardware
//
#include <hip/hip_runtime.h>
#include <hip/hip_bf16.h>

// CfC head: B=64, S=2048, H=1024. Sequential scan over S; elementwise
// recurrence per (b,h); per-step projection H->2 done as time-tiled WMMA.
// Trans-bound: gates use v_exp_f32 in log2 domain; sigmoids use v_tanh_f32.

#define Bsz 64
#define Ssz 2048
#define Hsz 1024
#define HB  256      // h-values per block (8 waves * 32)
#define TT  16       // time tile (WMMA M dimension)

typedef _Float16 v16h __attribute__((ext_vector_type(16)));
typedef _Float16 v8h  __attribute__((ext_vector_type(8)));
typedef float    v8f  __attribute__((ext_vector_type(8)));

#define L2E 1.4426950408889634f

// sigmoid(2z) given the pre-halved argument z: 0.5*tanh(z)+0.5
__device__ __forceinline__ float sig_half(float z) {
#if __has_builtin(__builtin_amdgcn_tanhf)
    return fmaf(0.5f, __builtin_amdgcn_tanhf(z), 0.5f);   // v_tanh_f32
#else
    return __builtin_amdgcn_rcpf(
        1.0f + __builtin_amdgcn_exp2f(z * (-2.0f * L2E)));
#endif
}

// Fill output with proj_b broadcast: out[b,s,j] = proj_b[j]
__global__ void cfc_init_out(float* __restrict__ out,
                             const float* __restrict__ proj_b, int n) {
    int i = blockIdx.x * blockDim.x + threadIdx.x;
    if (i < n) out[i] = proj_b[i & 1];
}

__global__ __launch_bounds__(HB) void cfc_scan_kernel(
    const float* __restrict__ x_codes,
    const float* __restrict__ Wi_w, const float* __restrict__ Wi_b,
    const float* __restrict__ Wf_w, const float* __restrict__ Wf_b,
    const float* __restrict__ Wo_w, const float* __restrict__ Wo_b,
    const float* __restrict__ Wg_w, const float* __restrict__ Wg_b,
    const float* __restrict__ Wl_w, const float* __restrict__ Wl_b,
    const float* __restrict__ proj_w,
    const float* __restrict__ n_init,
    float* __restrict__ out)
{
    __shared__ __align__(16) _Float16 hs[TT * HB];   // staged h (f16), 8 KB
    __shared__ float xt_s[TT];

    const int tid  = threadIdx.x;
    const int lane = tid & 31;
    const int wave = tid >> 5;
    const int b    = blockIdx.x >> 2;         // 4 blocks per batch
    const int hb   = (blockIdx.x & 3) * HB;   // this block's h offset
    const int h    = hb + tid;

    // Per-thread parameters, prescaled:
    //  - i/f/o gate weights folded with log2(e) (exp -> exp2, n in log2 dom)
    //  - g/lambda weights halved (sigmoid via tanh(x/2))
    const float wil = Wi_w[h] * L2E,  bil = Wi_b[h] * L2E;
    const float wfl = Wf_w[h] * L2E,  bfl = Wf_b[h] * L2E;
    const float wol = Wo_w[h] * L2E,  bol = Wo_b[h] * L2E;
    const float wgh = Wg_w[h] * 0.5f, bgh = Wg_b[h] * 0.5f;
    const float wlh = Wl_w[h] * 0.5f, blh = Wl_b[h] * 0.5f;
    const float C1  = 0.01f * L2E;    // n-update gain in log2 domain

    // State (registers); n tracked as nl = n * log2(e)
    float hst = 0.0f, cst = 0.0f, nl = n_init[h] * L2E;

    // Constant B fragment for this wave's 32-wide h chunk.
    // B 16-bit 32x16 layout: lane L holds column N=L&15;
    // element j = B[K = (L>=16?8:0)+j+(j>=8?8:0), N]. Only N in {0,1} nonzero.
    const int nsel  = lane & 15;
    const int kbase = (lane & 16) ? 8 : 0;
    v16h bfrag;
#pragma unroll
    for (int j = 0; j < 16; ++j) {
        float pv = 0.0f;
        if (nsel < 2) pv = proj_w[nsel * Hsz + hb + wave * 32 + kbase + j + ((j >> 3) << 3)];
        bfrag[j] = (_Float16)pv;
    }

    const float* xrow = x_codes + (size_t)b * Ssz;

    // A-fragment addressing (16-bit A 16x32 layout):
    // lane L: M = L&15; low 8 halves = K (L<16 ? 0..7 : 8..15),
    //                   high 8 halves = K (L<16 ? 16..23 : 24..31)
    const int am    = lane & 15;
    const int ahalf = (lane >> 4) & 1;
    const _Float16* abase = &hs[am * HB + wave * 32 + ahalf * 8];

    for (int s0 = 0; s0 < Ssz; s0 += TT) {
        if (tid < TT) xt_s[tid] = (xrow[s0 + tid] - 65.0f) * 0.01f;
        __syncthreads();   // x ready; prev-tile h_tile reads complete

#pragma unroll
        for (int t = 0; t < TT; ++t) {
            const float xt  = xt_s[t];
            // i/f/o gates: exp(pre - n) = exp2(fma(xt,w*l2e,b*l2e) - nl)
            const float it  = __builtin_amdgcn_exp2f(fmaf(xt, wil, bil) - nl);
            const float ft  = __builtin_amdgcn_exp2f(fmaf(xt, wfl, bfl) - nl);
            const float ot  = __builtin_amdgcn_exp2f(fmaf(xt, wol, bol) - nl);
            // sigmoids via tanh (weights pre-halved)
            const float gt  = sig_half(fmaf(xt, wgh, bgh));
            const float lam = sig_half(fmaf(xt, wlh, blh));
            cst = fmaf(ft, cst, it * gt);
            const float ff = ot * sig_half(0.5f * cst);
            nl  = fmaf(C1, (it + ft + ot) - 3.0f, nl);
            hst = fmaf(0.01f, ff, hst) *
                  __builtin_amdgcn_rcpf(fmaf(0.01f, lam, 1.0f));
            hs[t * HB + tid] = (_Float16)hst;
        }
        __syncthreads();   // h tile fully staged

        // Load A fragment (two 16B LDS vector loads per lane)
        const v8h alo = *(const v8h*)(abase);
        const v8h ahi = *(const v8h*)(abase + 16);
        const v16h afrag = __builtin_shufflevector(
            alo, ahi, 0, 1, 2, 3, 4, 5, 6, 7, 8, 9, 10, 11, 12, 13, 14, 15);

        v8f acc = {};
        acc = __builtin_amdgcn_wmma_f32_16x16x32_f16(
            /*neg_a=*/false, afrag, /*neg_b=*/false, bfrag,
            /*c_mod=*/(short)0, acc, /*reuse_a=*/false, /*reuse_b=*/false);

        // D 16x16 f32 layout: lane L, VGPR v -> M = v + (L>=16 ? 8 : 0),
        // N = L&15. Columns 0/1 are the two projections; M is time-in-tile.
        if (nsel < 2) {
            float* op = out + ((size_t)b * Ssz + s0 + ahalf * 8) * 2 + nsel;
#pragma unroll
            for (int v = 0; v < 8; ++v)
                unsafeAtomicAdd(op + 2 * v, acc[v]);
        }
    }
}

extern "C" void kernel_launch(void* const* d_in, const int* in_sizes, int n_in,
                              void* d_out, int out_size, void* d_ws, size_t ws_size,
                              hipStream_t stream) {
    const float* x_codes = (const float*)d_in[0];
    const float* Wi_w    = (const float*)d_in[1];
    const float* Wi_b    = (const float*)d_in[2];
    const float* Wf_w    = (const float*)d_in[3];
    const float* Wf_b    = (const float*)d_in[4];
    const float* Wo_w    = (const float*)d_in[5];
    const float* Wo_b    = (const float*)d_in[6];
    const float* Wg_w    = (const float*)d_in[7];
    const float* Wg_b    = (const float*)d_in[8];
    const float* Wl_w    = (const float*)d_in[9];
    const float* Wl_b    = (const float*)d_in[10];
    const float* proj_w  = (const float*)d_in[11];
    const float* proj_b  = (const float*)d_in[12];
    const float* n_init  = (const float*)d_in[13];
    float* out = (float*)d_out;

    const int n = Bsz * Ssz * 2;
    cfc_init_out<<<(n + 255) / 256, 256, 0, stream>>>(out, proj_b, n);

    // 4 blocks per batch -> 256 workgroups of 8 wave32s: spreads the
    // 65536 recurrence lanes across the WGP array.
    cfc_scan_kernel<<<dim3(Bsz * (Hsz / HB)), dim3(HB), 0, stream>>>(
        x_codes, Wi_w, Wi_b, Wf_w, Wf_b, Wo_w, Wo_b, Wg_w, Wg_b,
        Wl_w, Wl_b, proj_w, n_init, out);
}
